// RNN_90263032693240
// MI455X (gfx1250) — matrix operand order repaired and checked by reading, not compiled, not run on hardware
//
#include <hip/hip_runtime.h>
#include <hip/hip_bf16.h>

// ---------------------------------------------------------------------------
// BiLSTM-CRF for MI455X (gfx1250, wave32, WMMA).
//
// Pipeline (all on `stream`):
//   1. convert weights f32 -> bf16 (w_ih / w_hh / lin_w)
//   2. embedding gather -> x_bf16 [T,256]
//   3. pregate GEMM (WMMA bf16):  PG = X @ W_ih^T + (b_ih + b_hh)   [T,2048]
//   4. recurrent kernel (WMMA bf16 matvec, M=16 padded, h/c in LDS,
//      global_prefetch of next step's pregates) -> h bf16
//   5. repeat 3+4 for layer 1 (input = concat(h_fwd, h_bwd) [T,1024])
//   6. linear feats via WMMA [T,48], CRF forward scan -> scalar loss
// ---------------------------------------------------------------------------

typedef __attribute__((ext_vector_type(16))) __bf16 v16bf;
typedef __attribute__((ext_vector_type(8)))  __bf16 v8bf;
typedef __attribute__((ext_vector_type(8)))  float  v8f;

#define T_SEQ 2048
#define HID   512
#define GATES 2048   // 4*HID
#define EMB   256
#define NSTATE 48
#define START_TAG 45
#define END_TAG   46
#define NEG_INF (-100000.0f)

// ---------------------------------------------------------------------------
// WMMA fragment helpers (wave32; this code's contiguous k order, used alike for A and B).
//
// 16-bit A matrix (16x32, M x K): lane L holds row M = L&15; halves are a
// contiguous 16-element run over K at kb + (L>=16 ? 16:0) (two 16B loads).
//
// 16-bit B matrix (32x16, K x N): lane L holds column N = L&15; halves are a
// contiguous 16-element (32B) run over K at kb + (L>=16 ? 16:0).  With the B
// operand being W^T and W row-major [N, Kdim], the run is contiguous in W.
// ---------------------------------------------------------------------------

__device__ __forceinline__ v16bf load_a_frag(const __bf16* __restrict__ rowp,
                                             int kb, int lane) {
  const __bf16* p0 = rowp + kb + ((lane & 16) ? 16 : 0);
  v8bf lo = *(const v8bf*)(p0);
  v8bf hi = *(const v8bf*)(p0 + 8);
  v16bf a;
#pragma unroll
  for (int i = 0; i < 8; ++i) { a[i] = lo[i]; a[i + 8] = hi[i]; }
  return a;
}

__device__ __forceinline__ v8f wmma_bf16(v16bf a, v16bf b, v8f c) {
  // (neg_a, A, neg_b, B, c_mod, C, reuse_a, reuse_b)
  return __builtin_amdgcn_wmma_f32_16x16x32_bf16(false, a, false, b,
                                                 (short)0, c, false, false);
}

__device__ __forceinline__ float sigf(float x) {
  return 1.0f / (1.0f + __expf(-x));
}

// ---------------------------------------------------------------------------
// 1. f32 -> bf16 conversion (weights). n must be a multiple of 256.
// ---------------------------------------------------------------------------
__global__ void cvt_bf16_kernel(__bf16* __restrict__ dst,
                                const float* __restrict__ src, int n) {
  int i = blockIdx.x * blockDim.x + threadIdx.x;
  if (i < n) dst[i] = (__bf16)src[i];
}

// ---------------------------------------------------------------------------
// 2. Embedding gather + convert: x_bf[t, e] = embed[tokens[t], e]
// ---------------------------------------------------------------------------
__global__ void embed_kernel(__bf16* __restrict__ xbf,
                             const float* __restrict__ embed,
                             const int* __restrict__ tokens) {
  int t = blockIdx.x;
  int e = threadIdx.x;               // 256 threads
  int tok = tokens[t];
  xbf[(size_t)t * EMB + e] = (__bf16)embed[(size_t)tok * EMB + e];
}

// ---------------------------------------------------------------------------
// 3. Pregate GEMM:  PG[m, n] = sum_k X[m,k] * W[n,k] + b_ih[n] + b_hh[n]
//    X: [T, Kdim] bf16 row-major,  W: [2048, Kdim] bf16 row-major (= W_ih).
//    One wave per 16x16 output tile; 4 waves per block cover 4 N-tiles.
// ---------------------------------------------------------------------------
__global__ __launch_bounds__(128)
void pregate_gemm_kernel(float* __restrict__ pg,
                         const __bf16* __restrict__ X,
                         const __bf16* __restrict__ W,
                         const float* __restrict__ b_ih,
                         const float* __restrict__ b_hh,
                         int Kdim) {
  const int lane  = threadIdx.x;                       // 0..31
  const int ntile = blockIdx.x * 4 + threadIdx.y;      // 0..127
  const int mtile = blockIdx.y;                        // 0..127
  const int m0 = mtile * 16, n0 = ntile * 16;

  const __bf16* arow = X + (size_t)(m0 + (lane & 15)) * Kdim;
  const __bf16* bptr = W + (size_t)(n0 + (lane & 15)) * Kdim
                         + ((lane & 16) ? 16 : 0);
  v8f acc = {};
  for (int kb = 0; kb < Kdim; kb += 32) {
    v16bf a = load_a_frag(arow, kb, lane);
    v16bf b = *(const v16bf*)(bptr + kb);
    acc = wmma_bf16(a, b, acc);
  }
  // C/D layout: element r of acc -> row m0 + r + (lane>=16 ? 8:0), col n0+(lane&15)
  const int n = n0 + (lane & 15);
  const float bias = b_ih[n] + b_hh[n];
  const int mbase = m0 + ((lane & 16) ? 8 : 0);
#pragma unroll
  for (int r = 0; r < 8; ++r)
    pg[(size_t)(mbase + r) * GATES + n] = acc[r] + bias;
}

// ---------------------------------------------------------------------------
// 4. Recurrent LSTM layer. grid.x = 2 (dir 0 = forward, dir 1 = backward),
//    1024 threads = 32 waves (max workgroup = 8 waves/SIMD32 on the WGP).
//    Per step: g = PG[t] + h_prev @ W_hh^T via a row-replicated M=16 WMMA
//    matvec (row 0 of D is the real result), then the pointwise cell update.
//    h (bf16) and c (f32) live in LDS; W_hh streams from L2 each step.
//    Next step's pregate row is prefetched behind the WMMAs.
//    h_out: [T, 1024] bf16; this direction writes columns dir*512 ..
// ---------------------------------------------------------------------------
__global__ __launch_bounds__(1024)
void lstm_layer_kernel(__bf16* __restrict__ h_out,
                       const float* __restrict__ pg_f,
                       const float* __restrict__ pg_b,
                       const __bf16* __restrict__ whh_f,
                       const __bf16* __restrict__ whh_b) {
  const int dir = blockIdx.x;
  const float*  pg  = dir ? pg_b  : pg_f;
  const __bf16* whh = dir ? whh_b : whh_f;

  __shared__ __attribute__((aligned(32))) __bf16 h_bf[HID];
  __shared__ float gates[GATES];
  __shared__ float cst[HID];

  const int tid  = threadIdx.x;
  const int lane = tid & 31;
  const int wave = tid >> 5;     // 0..31

  if (tid < HID) { h_bf[tid] = (__bf16)0.0f; cst[tid] = 0.0f; }
  __syncthreads();

  for (int step = 0; step < T_SEQ; ++step) {
    const int t = dir ? (T_SEQ - 1 - step) : step;

    // Prefetch next step's pregate row (8 KB) into cache behind the WMMAs.
    if (step + 1 < T_SEQ) {
      const int tn = dir ? (t - 1) : (t + 1);
      __builtin_prefetch(&pg[(size_t)tn * GATES + tid * 2], 0, 3);
    }

    // 32 waves x 4 tiles x 16 cols = 2048 gate pre-activations.
#pragma unroll 1
    for (int ti = 0; ti < 4; ++ti) {
      const int n0 = (wave * 4 + ti) * 16;
      const __bf16* bptr = whh + (size_t)(n0 + (lane & 15)) * HID
                               + ((lane & 16) ? 16 : 0);
      v8f acc = {};
#pragma unroll
      for (int kb = 0; kb < HID; kb += 32) {
        v16bf a = load_a_frag(h_bf, kb, lane);      // row-replicated h (LDS)
        v16bf b = *(const v16bf*)(bptr + kb);       // W_hh slice (L2 resident)
        acc = wmma_bf16(a, b, acc);
      }
      if (lane < 16) {                              // row 0 = real matvec result
        const int n = n0 + lane;
        gates[n] = acc[0] + pg[(size_t)t * GATES + n];
      }
    }
    __syncthreads();

    if (tid < HID) {                                // pointwise LSTM cell
      const float gi = gates[tid];
      const float gf = gates[HID + tid];
      const float gg = gates[2 * HID + tid];
      const float go = gates[3 * HID + tid];
      const float c  = sigf(gf) * cst[tid] + sigf(gi) * tanhf(gg);
      cst[tid] = c;
      const float h = sigf(go) * tanhf(c);
      const __bf16 hb = (__bf16)h;
      h_bf[tid] = hb;
      h_out[(size_t)t * (2 * HID) + dir * HID + tid] = hb;
    }
    __syncthreads();
  }
}

// ---------------------------------------------------------------------------
// 5. Final linear via WMMA: feats[m, n] = sum_k h1[m,k] * lin_w[n,k] + lin_b[n]
//    h1: [T,1024] bf16, lin_w(bf16): [48,1024], feats: [T,48] f32.
//    Grid (3 N-tiles, 128 M-tiles), one wave per 16x16 tile, K = 1024.
// ---------------------------------------------------------------------------
__global__ __launch_bounds__(32)
void linear_wmma_kernel(float* __restrict__ feats,
                        const __bf16* __restrict__ h1,
                        const __bf16* __restrict__ lwb,
                        const float* __restrict__ lb) {
  const int lane = threadIdx.x;
  const int n0 = blockIdx.x * 16;      // 0, 16, 32
  const int m0 = blockIdx.y * 16;
  const int Kdim = 2 * HID;

  const __bf16* arow = h1 + (size_t)(m0 + (lane & 15)) * Kdim;
  const __bf16* bptr = lwb + (size_t)(n0 + (lane & 15)) * Kdim
                           + ((lane & 16) ? 16 : 0);
  v8f acc = {};
  for (int kb = 0; kb < Kdim; kb += 32) {
    v16bf a = load_a_frag(arow, kb, lane);
    v16bf b = *(const v16bf*)(bptr + kb);
    acc = wmma_bf16(a, b, acc);
  }
  const int n = n0 + (lane & 15);
  const float bias = lb[n];
  const int mbase = m0 + ((lane & 16) ? 8 : 0);
#pragma unroll
  for (int r = 0; r < 8; ++r)
    feats[(size_t)(mbase + r) * NSTATE + n] = acc[r] + bias;
}

// ---------------------------------------------------------------------------
// 6. CRF forward scan + gold score -> loss.  One block, 64 threads (2 waves).
// ---------------------------------------------------------------------------
__global__ void crf_kernel(float* __restrict__ out,
                           const float* __restrict__ feats,
                           const float* __restrict__ trans,
                           const int* __restrict__ tags,
                           const int* __restrict__ seq_len) {
  __shared__ float alpha[NSTATE];
  __shared__ float nalpha[NSTATE];
  const int j = threadIdx.x;
  if (j < NSTATE) alpha[j] = (j == START_TAG) ? 0.0f : NEG_INF;
  __syncthreads();

  for (int t = 0; t < T_SEQ; ++t) {
    if (j < NSTATE) {
      const float* tr = trans + j * NSTATE;
      float m = -3.4e38f;
#pragma unroll 1
      for (int i = 0; i < NSTATE; ++i) m = fmaxf(m, alpha[i] + tr[i]);
      float s = 0.0f;
#pragma unroll 1
      for (int i = 0; i < NSTATE; ++i) s += __expf(alpha[i] + tr[i] - m);
      nalpha[j] = m + __logf(s) + feats[t * NSTATE + j];
    }
    __syncthreads();
    if (j < NSTATE) alpha[j] = nalpha[j];
    __syncthreads();
  }

  if (j == 0) {
    // log Z = logsumexp(alpha + trans[END, :])
    float m = -3.4e38f;
    for (int i = 0; i < NSTATE; ++i)
      m = fmaxf(m, alpha[i] + trans[END_TAG * NSTATE + i]);
    float s = 0.0f;
    for (int i = 0; i < NSTATE; ++i)
      s += __expf(alpha[i] + trans[END_TAG * NSTATE + i] - m);
    const float log_z = m + __logf(s);

    // gold score
    int prev = START_TAG;
    float score = 0.0f;
    for (int t = 0; t < T_SEQ; ++t) {
      const int cur = tags[t];
      score += trans[cur * NSTATE + prev] + feats[t * NSTATE + cur];
      prev = cur;
    }
    score += trans[END_TAG * NSTATE + prev];
    out[0] = (log_z - score) / (float)seq_len[0];
  }
}

// ---------------------------------------------------------------------------
// Orchestration
// ---------------------------------------------------------------------------
static inline size_t align256(size_t x) { return (x + 255) & ~(size_t)255; }

extern "C" void kernel_launch(void* const* d_in, const int* in_sizes, int n_in,
                              void* d_out, int out_size, void* d_ws, size_t ws_size,
                              hipStream_t stream) {
  (void)in_sizes; (void)n_in; (void)out_size; (void)ws_size;

  const int*   tokens = (const int*)  d_in[0];
  const int*   tags   = (const int*)  d_in[1];
  const int*   slen   = (const int*)  d_in[2];
  const float* embed  = (const float*)d_in[3];
  // per (layer, dir): w_ih, w_hh, b_ih, b_hh
  const float* w_ih[2][2] = {{(const float*)d_in[4],  (const float*)d_in[8]},
                             {(const float*)d_in[12], (const float*)d_in[16]}};
  const float* w_hh[2][2] = {{(const float*)d_in[5],  (const float*)d_in[9]},
                             {(const float*)d_in[13], (const float*)d_in[17]}};
  const float* b_ih[2][2] = {{(const float*)d_in[6],  (const float*)d_in[10]},
                             {(const float*)d_in[14], (const float*)d_in[18]}};
  const float* b_hh[2][2] = {{(const float*)d_in[7],  (const float*)d_in[11]},
                             {(const float*)d_in[15], (const float*)d_in[19]}};
  const float* lin_w = (const float*)d_in[20];
  const float* lin_b = (const float*)d_in[21];
  const float* trans = (const float*)d_in[22];
  float* out = (float*)d_out;

  // ---- workspace carve ----
  char* ws = (char*)d_ws;
  size_t off = 0;
  auto carve = [&](size_t bytes) { char* p = ws + off; off += align256(bytes); return p; };

  __bf16* xbf   = (__bf16*)carve((size_t)T_SEQ * EMB * 2);            // [T,256]
  __bf16* h0bf  = (__bf16*)carve((size_t)T_SEQ * 2 * HID * 2);        // [T,1024]
  __bf16* h1bf  = (__bf16*)carve((size_t)T_SEQ * 2 * HID * 2);        // [T,1024]
  __bf16* wihb[2][2];
  __bf16* whhb[2][2];
  const int kin[2] = {EMB, 2 * HID};
  for (int l = 0; l < 2; ++l)
    for (int d = 0; d < 2; ++d) {
      wihb[l][d] = (__bf16*)carve((size_t)GATES * kin[l] * 2);
      whhb[l][d] = (__bf16*)carve((size_t)GATES * HID * 2);
    }
  __bf16* lwb  = (__bf16*)carve((size_t)NSTATE * 2 * HID * 2);        // [48,1024]
  float* pg_f  = (float*)carve((size_t)T_SEQ * GATES * 4);            // [T,2048]
  float* pg_b  = (float*)carve((size_t)T_SEQ * GATES * 4);
  float* feats = (float*)carve((size_t)T_SEQ * NSTATE * 4);

  // ---- 1. weight conversion ----
  for (int l = 0; l < 2; ++l)
    for (int d = 0; d < 2; ++d) {
      int n1 = GATES * kin[l];
      cvt_bf16_kernel<<<n1 / 256, 256, 0, stream>>>(wihb[l][d], w_ih[l][d], n1);
      int n2 = GATES * HID;
      cvt_bf16_kernel<<<n2 / 256, 256, 0, stream>>>(whhb[l][d], w_hh[l][d], n2);
    }
  cvt_bf16_kernel<<<(NSTATE * 2 * HID) / 256, 256, 0, stream>>>(
      lwb, lin_w, NSTATE * 2 * HID);

  // ---- 2. embedding ----
  embed_kernel<<<T_SEQ, EMB, 0, stream>>>(xbf, embed, tokens);

  const dim3 gemm_block(32, 4);
  const dim3 gemm_grid(GATES / 16 / 4, T_SEQ / 16);   // (32, 128)

  // ---- 3/4. layer 0 ----
  pregate_gemm_kernel<<<gemm_grid, gemm_block, 0, stream>>>(
      pg_f, xbf, wihb[0][0], b_ih[0][0], b_hh[0][0], EMB);
  pregate_gemm_kernel<<<gemm_grid, gemm_block, 0, stream>>>(
      pg_b, xbf, wihb[0][1], b_ih[0][1], b_hh[0][1], EMB);
  lstm_layer_kernel<<<2, 1024, 0, stream>>>(h0bf, pg_f, pg_b,
                                            whhb[0][0], whhb[0][1]);

  // ---- 3/4. layer 1 ----
  pregate_gemm_kernel<<<gemm_grid, gemm_block, 0, stream>>>(
      pg_f, h0bf, wihb[1][0], b_ih[1][0], b_hh[1][0], 2 * HID);
  pregate_gemm_kernel<<<gemm_grid, gemm_block, 0, stream>>>(
      pg_b, h0bf, wihb[1][1], b_ih[1][1], b_hh[1][1], 2 * HID);
  lstm_layer_kernel<<<2, 1024, 0, stream>>>(h1bf, pg_f, pg_b,
                                            whhb[1][0], whhb[1][1]);

  // ---- 5. linear (WMMA) ----
  linear_wmma_kernel<<<dim3(NSTATE / 16, T_SEQ / 16), 32, 0, stream>>>(
      feats, h1bf, lwb, lin_b);

  // ---- 6. CRF ----
  crf_kernel<<<1, 64, 0, stream>>>(out, feats, trans, tags, slen);
}
